// SRNN_57741540327965
// MI455X (gfx1250) — matrix operand order repaired and checked by reading, not compiled, not run on hardware
//
#include <hip/hip_runtime.h>
#include <math.h>

typedef __bf16 bf16_t;
typedef __bf16 v16bf __attribute__((ext_vector_type(16)));
typedef __bf16 v8bf  __attribute__((ext_vector_type(8)));
typedef float  v8f   __attribute__((ext_vector_type(8)));

// Global-address-space pointer types: keep AS(1) in the TYPE so that pointer
// laundering (asm) cannot destroy the address-space provenance -> loads stay
// global_load_b128 with saddr + immediate offset.
typedef const bf16_t __attribute__((address_space(1)))* gcbf16p;
typedef const v16bf  __attribute__((address_space(1)))* gcv16p;

#define T_STEPS 12
#define NNODES  128
#define ERNN    192
#define G3      576
#define NRNN    128

// ---------------------------------------------------------------------------
// Shared memory block layout: 64,000 bytes total.
// ---------------------------------------------------------------------------
struct SMem {
  bf16_t hbf[NNODES * ERNN];   // 49152 B  bf16 shadow of h_edge rows (A operands)
  float  part[8 * ERNN];       //  6144 B  per-wave spatial-sum partials
  float  hsp[ERNN];            //   768 B  spatial sum per column
  float  ht[2][ERNN];          //  1536 B  temporal edge hidden (double buffered)
  float  et[64];               //   256 B  temporal edge encoding
  float  encp[64];             //   256 B  node position encoding
  float  g1[G3];               //  2304 B  matvec scratch (gi)
  float  g2[G3];               //  2304 B  matvec scratch (gh)
  float  hn[2][NRNN];          //  1024 B  node hidden (double buffered)
  float  ench[64];             //   256 B  node hidden encoding
};

__device__ __forceinline__ float sigm(float x) { return 1.0f / (1.0f + __expf(-x)); }

__device__ __forceinline__ v8f wmma_bf16(v16bf a, v16bf b, v8f c) {
  // D = A(16x32 bf16) * B(32x16 bf16) + C(16x16 f32)
  return __builtin_amdgcn_wmma_f32_16x16x32_bf16(false, a, false, b, (short)0, c,
                                                 false, false);
}

// B fragment, fragment-major in global scratch: tile (kt,nt) is 512 bf16,
// lane-contiguous 16 values per lane. Base is a laundered AS(1) pointer, so
// the constant tile index folds into the 24-bit instruction offset and the
// only VGPR address component is the shared lane*32B term.
__device__ __forceinline__ v16bf load_bfrag(gcbf16p frag, int kt, int nt, int lane) {
  return *(gcv16p)(frag + (((kt * 36 + nt) << 9) + (lane << 4)));
}

// A fragment from row-major bf16 LDS (stride elements). Per ISA 16-bit A
// layout: lane<16 holds K {kb..kb+7, kb+16..kb+23} with kb=ktile*32,
// lane>=16 the same shifted by 8. Two contiguous 16B runs per lane.
__device__ __forceinline__ v16bf load_afrag(const bf16_t* base, int stride, int m, int kb) {
  union { v16bf v; v8bf h[2]; } u;
  const bf16_t* p = base + m * stride + kb;
  u.h[0] = *(const v8bf*)p;
  u.h[1] = *(const v8bf*)(p + 16);
  return u.v;
}

// ---------------------------------------------------------------------------
// Weight prep: fp32 row-major W[K][576] -> bf16 fragment-major tiles.
// Dense 16-bit B layout: lane = N within tile, k_local = (lane>=16)*8 + (i<8 ? i : i+8).
// ---------------------------------------------------------------------------
__global__ void srnn_prep_frag(const float* __restrict__ W, bf16_t* __restrict__ frag,
                               int KT) {
  int e = blockIdx.x * 256 + threadIdx.x;
  int total = KT * 36 * 512;
  if (e >= total) return;
  int tile = e >> 9;
  int r    = e & 511;
  int lane = r >> 4;
  int i    = r & 15;
  int kt = tile / 36, nt = tile % 36;
  int k = kt * 32 + ((lane >> 4) << 3) + (i < 8 ? i : i + 8);
  int n = nt * 16 + (lane & 15);
  frag[e] = (bf16_t)W[k * G3 + n];
}

// ---------------------------------------------------------------------------
// Fused SRNN scan: one block per node, persistent over all T steps.
// ---------------------------------------------------------------------------
__global__ void __launch_bounds__(256, 1)
srnn_scan(const float* __restrict__ nodes,      // [T,128,2]
          const float* __restrict__ edges,      // [T,16384,2]
          const float* __restrict__ hnode0,     // [128,128]
          const float* __restrict__ hedge0,     // [16384,192]
          const float* __restrict__ Wt_enc, const float* __restrict__ bt_enc,
          const float* __restrict__ Wt_ih,  const float* __restrict__ Wt_hh,
          const float* __restrict__ bt_ih,  const float* __restrict__ bt_hh,
          const float* __restrict__ Ws_enc, const float* __restrict__ bs_enc,
          const float* __restrict__ bs_ih,  const float* __restrict__ bs_hh,
          const float* __restrict__ Wn_enc, const float* __restrict__ bn_enc,
          const float* __restrict__ Wn_henc,const float* __restrict__ bn_henc,
          const float* __restrict__ Wn_ih,  const float* __restrict__ Wn_hh,
          const float* __restrict__ bn_ih,  const float* __restrict__ bn_hh,
          const float* __restrict__ Wn_dec, const float* __restrict__ bn_dec,
          const bf16_t* __restrict__ fih,   // Ws_ih fragments [2][36][512]
          const bf16_t* __restrict__ fhh,   // Ws_hh fragments [6][36][512]
          float* __restrict__ out) {
  __shared__ SMem sm;

  const int b    = blockIdx.x;       // node id / edge-row base b*128
  const int tid  = threadIdx.x;
  const int wid  = tid >> 5;         // wave id == M-tile
  const int lane = tid & 31;
  const int ln15 = lane & 15;
  const int hi   = lane >> 4;        // 0/1 half of the wave

  // Per-lane slice of the C/D fragment layout:
  //   rows m = wid*16 + hi*8 + v (v=0..7), column c = j*16 + ln15 (j=0..11).
  float hreg[12][8];

  // ---- init: load h_edge rows into regs + bf16 LDS shadow ----
  {
    const int mbase = wid * 16 + hi * 8;
#pragma unroll
    for (int j = 0; j < 12; ++j) {
      const int c = j * 16 + ln15;
#pragma unroll
      for (int v = 0; v < 8; ++v) {
        float h0 = hedge0[((size_t)(b * NNODES + mbase + v)) * ERNN + c];
        hreg[j][v] = h0;
        sm.hbf[(mbase + v) * ERNN + c] = (bf16_t)h0;
      }
    }
  }
  if (tid < ERNN) sm.ht[0][tid] = hedge0[((size_t)(b * NNODES + b)) * ERNN + tid];
  if (tid < NRNN) sm.hn[0][tid] = hnode0[b * NRNN + tid];
  __syncthreads();

  int cur = 0;
#pragma unroll 1
  for (int t = 0; t < T_STEPS; ++t) {
    const int nxt = cur ^ 1;
    const float* edg = edges + ((size_t)t * NNODES * NNODES + (size_t)b * NNODES) * 2;
    const float* nd  = nodes + ((size_t)t * NNODES + b) * 2;

    // ---- phase 1: encoders ----
    // e_s A-fragments built straight into registers (lane owns row m = wid*16+ln15,
    // its K-subset per the 16-bit A fragment layout).
    v16bf agi[2];
    {
      const int m = wid * 16 + ln15;
      const float x0 = edg[2 * m], x1 = edg[2 * m + 1];
#pragma unroll
      for (int kt = 0; kt < 2; ++kt) {
#pragma unroll
        for (int i = 0; i < 16; ++i) {
          const int k = kt * 32 + hi * 8 + (i < 8 ? i : i + 8);
          float v = fmaf(x1, Ws_enc[64 + k], fmaf(x0, Ws_enc[k], bs_enc[k]));
          agi[kt][i] = (bf16_t)fmaxf(v, 0.0f);
        }
      }
    }
    if (tid < 64) {                     // temporal edge encoding (diagonal edge)
      const float x0 = edg[2 * b], x1 = edg[2 * b + 1];
      float v = fmaf(x1, Wt_enc[64 + tid], fmaf(x0, Wt_enc[tid], bt_enc[tid]));
      sm.et[tid] = fmaxf(v, 0.0f);
    } else if (tid < 128) {             // node position encoding
      const int c = tid - 64;
      const float x0 = nd[0], x1 = nd[1];
      float v = fmaf(x1, Wn_enc[64 + c], fmaf(x0, Wn_enc[c], bn_enc[c]));
      sm.encp[c] = fmaxf(v, 0.0f);
    }
    __syncthreads();

    // ---- phase 2: spatial GRU via WMMA (wave = M-tile, 4 column chunks of 3
    // tiles each; accumulators = 12 x v8f = 96 VGPRs live). ----
#pragma unroll
    for (int chunk = 0; chunk < 4; ++chunk) {
      // Launder the AS(1)-typed fragment base pointers every chunk: the asm
      // defeats LICM (no hoist-and-spill of the t-invariant weight stream),
      // while the address space survives in the pointer TYPE, so loads lower
      // to global_load_b128 [saddr + lane*32B vaddr + imm offset].
      gcbf16p fihp = (gcbf16p)fih;
      gcbf16p fhhp = (gcbf16p)fhh;
      asm volatile("" : "+s"(fihp), "+s"(fhhp));

      v8f accR[3], accZ[3], accI[3], accH[3];
#pragma unroll
      for (int q = 0; q < 3; ++q) {
        accR[q] = (v8f){0,0,0,0,0,0,0,0};
        accZ[q] = (v8f){0,0,0,0,0,0,0,0};
        accI[q] = (v8f){0,0,0,0,0,0,0,0};
        accH[q] = (v8f){0,0,0,0,0,0,0,0};
      }
      // gi = e_s @ Ws_ih  (K=64): r,z accumulate shared with gh; inew separate
#pragma unroll
      for (int kt = 0; kt < 2; ++kt) {
        const v16bf a = agi[kt];
#pragma unroll
        for (int q = 0; q < 3; ++q) {
          const int nt = chunk * 3 + q;
          accR[q] = wmma_bf16(a, load_bfrag(fihp, kt, nt,      lane), accR[q]);
          accZ[q] = wmma_bf16(a, load_bfrag(fihp, kt, 12 + nt, lane), accZ[q]);
          accI[q] = wmma_bf16(a, load_bfrag(fihp, kt, 24 + nt, lane), accI[q]);
        }
      }
      // gh = h_edge @ Ws_hh (K=192), A from bf16 LDS shadow
#pragma unroll
      for (int kt = 0; kt < 6; ++kt) {
        const v16bf a = load_afrag(sm.hbf, ERNN, wid * 16 + ln15, kt * 32 + hi * 8);
#pragma unroll
        for (int q = 0; q < 3; ++q) {
          const int nt = chunk * 3 + q;
          accR[q] = wmma_bf16(a, load_bfrag(fhhp, kt, nt,      lane), accR[q]);
          accZ[q] = wmma_bf16(a, load_bfrag(fhhp, kt, 12 + nt, lane), accZ[q]);
          accH[q] = wmma_bf16(a, load_bfrag(fhhp, kt, 24 + nt, lane), accH[q]);
        }
      }
      // pointwise GRU update (h_old stays fp32 in registers)
#pragma unroll
      for (int q = 0; q < 3; ++q) {
        const int j = chunk * 3 + q;
        const int c = j * 16 + ln15;
        const float br = bs_ih[c] + bs_hh[c];
        const float bz = bs_ih[ERNN + c] + bs_hh[ERNN + c];
        const float bi = bs_ih[2 * ERNN + c];
        const float bh = bs_hh[2 * ERNN + c];
#pragma unroll
        for (int v = 0; v < 8; ++v) {
          const float r  = sigm(accR[q][v] + br);
          const float z  = sigm(accZ[q][v] + bz);
          const float nn = tanhf(accI[q][v] + bi + r * (accH[q][v] + bh));
          hreg[j][v] = (1.0f - z) * nn + z * hreg[j][v];
        }
      }
    }
    __syncthreads();  // all LDS A-fragment reads of old h done

    // ---- phase 3a: deterministic spatial sum (exclude diagonal row b) ----
#pragma unroll
    for (int j = 0; j < 12; ++j) {
      float p = 0.0f;
#pragma unroll
      for (int v = 0; v < 8; ++v) {
        const int m = wid * 16 + hi * 8 + v;
        p += (m == b) ? 0.0f : hreg[j][v];
      }
      p += __shfl_xor(p, 16, 32);       // combine the two half-wave partials
      if (hi == 0) sm.part[wid * ERNN + j * 16 + ln15] = p;
    }
    __syncthreads();
    if (tid < ERNN) {
      float s = 0.0f;
#pragma unroll
      for (int w = 0; w < 8; ++w) s += sm.part[w * ERNN + tid];
      sm.hsp[tid] = s;
    }

    // ---- phase 3b: temporal edge GRU matvecs (tiny, VALU) ----
    for (int c = tid; c < G3; c += 256) {
      float gi = bt_ih[c];
      for (int k = 0; k < 64; ++k)   gi = fmaf(sm.et[k], Wt_ih[k * G3 + c], gi);
      float gh = bt_hh[c];
      for (int k = 0; k < ERNN; ++k) gh = fmaf(sm.ht[cur][k], Wt_hh[k * G3 + c], gh);
      sm.g1[c] = gi; sm.g2[c] = gh;
    }
    __syncthreads();
    if (tid < ERNN) {
      const float r  = sigm(sm.g1[tid] + sm.g2[tid]);
      const float z  = sigm(sm.g1[ERNN + tid] + sm.g2[ERNN + tid]);
      const float nn = tanhf(sm.g1[2 * ERNN + tid] + r * sm.g2[2 * ERNN + tid]);
      sm.ht[nxt][tid] = (1.0f - z) * nn + z * sm.ht[cur][tid];
    }
    __syncthreads();

    // ---- phase 3c: diagonal fixup + refresh bf16 shadow for next step ----
    {
      const int mbase = wid * 16 + hi * 8;
#pragma unroll
      for (int j = 0; j < 12; ++j) {
        const int c = j * 16 + ln15;
#pragma unroll
        for (int v = 0; v < 8; ++v) {
          if (mbase + v == b) hreg[j][v] = sm.ht[nxt][c];
          sm.hbf[(mbase + v) * ERNN + c] = (bf16_t)hreg[j][v];
        }
      }
    }
    // node hidden-encoder: h_other = [h_t, h_spatial]
    if (tid < 64) {
      float a = bn_henc[tid];
      for (int k = 0; k < ERNN; ++k) a = fmaf(sm.ht[nxt][k], Wn_henc[k * 64 + tid], a);
      for (int k = 0; k < ERNN; ++k) a = fmaf(sm.hsp[k], Wn_henc[(ERNN + k) * 64 + tid], a);
      sm.ench[tid] = fmaxf(a, 0.0f);
    }
    __syncthreads();

    // ---- phase 4: node GRU (x = [enc_pos, enc_hid]) ----
    for (int c = tid; c < 3 * NRNN; c += 256) {
      float gi = bn_ih[c];
      for (int k = 0; k < 64; ++k) gi = fmaf(sm.encp[k], Wn_ih[k * 384 + c], gi);
      for (int k = 0; k < 64; ++k) gi = fmaf(sm.ench[k], Wn_ih[(64 + k) * 384 + c], gi);
      float gh = bn_hh[c];
      for (int k = 0; k < NRNN; ++k) gh = fmaf(sm.hn[cur][k], Wn_hh[k * 384 + c], gh);
      sm.g1[c] = gi; sm.g2[c] = gh;
    }
    __syncthreads();
    if (tid < NRNN) {
      const float r  = sigm(sm.g1[tid] + sm.g2[tid]);
      const float z  = sigm(sm.g1[NRNN + tid] + sm.g2[NRNN + tid]);
      const float nn = tanhf(sm.g1[2 * NRNN + tid] + r * sm.g2[2 * NRNN + tid]);
      sm.hn[nxt][tid] = (1.0f - z) * nn + z * sm.hn[cur][tid];
    }
    __syncthreads();
    if (tid < 5) {
      float a = bn_dec[tid];
      for (int k = 0; k < NRNN; ++k) a = fmaf(sm.hn[nxt][k], Wn_dec[k * 5 + tid], a);
      out[((size_t)t * NNODES + b) * 5 + tid] = a;
    }
    if (t + 1 < T_STEPS) {  // global_prefetch_b8 of next step's edge rows
      __builtin_prefetch(edges + ((size_t)(t + 1) * NNODES * NNODES + (size_t)b * NNODES) * 2, 0, 1);
    }
    cur ^= 1;
    __syncthreads();
  }

  // ---- finalize: h_node and h_edge state ----
  float* out_hn = out + (size_t)T_STEPS * NNODES * 5;       // [128,128]
  float* out_he = out_hn + (size_t)NNODES * NRNN;           // [16384,192]
  if (tid < NRNN) out_hn[b * NRNN + tid] = sm.hn[cur][tid];
  {
    const int mbase = wid * 16 + hi * 8;
#pragma unroll
    for (int j = 0; j < 12; ++j) {
      const int c = j * 16 + ln15;
#pragma unroll
      for (int v = 0; v < 8; ++v)
        out_he[((size_t)(b * NNODES + mbase + v)) * ERNN + c] = hreg[j][v];
    }
  }
}

extern "C" void kernel_launch(void* const* d_in, const int* in_sizes, int n_in,
                              void* d_out, int out_size, void* d_ws, size_t ws_size,
                              hipStream_t stream) {
  const float* nodes   = (const float*)d_in[0];
  const float* edges   = (const float*)d_in[1];
  const float* hnode0  = (const float*)d_in[2];
  const float* hedge0  = (const float*)d_in[3];
  const float* Wt_enc  = (const float*)d_in[4];
  const float* bt_enc  = (const float*)d_in[5];
  const float* Wt_ih   = (const float*)d_in[6];
  const float* Wt_hh   = (const float*)d_in[7];
  const float* bt_ih   = (const float*)d_in[8];
  const float* bt_hh   = (const float*)d_in[9];
  const float* Ws_enc  = (const float*)d_in[10];
  const float* bs_enc  = (const float*)d_in[11];
  const float* Ws_ih   = (const float*)d_in[12];
  const float* Ws_hh   = (const float*)d_in[13];
  const float* bs_ih   = (const float*)d_in[14];
  const float* bs_hh   = (const float*)d_in[15];
  const float* Wn_enc  = (const float*)d_in[16];
  const float* bn_enc  = (const float*)d_in[17];
  const float* Wn_henc = (const float*)d_in[18];
  const float* bn_henc = (const float*)d_in[19];
  const float* Wn_ih   = (const float*)d_in[20];
  const float* Wn_hh   = (const float*)d_in[21];
  const float* bn_ih   = (const float*)d_in[22];
  const float* bn_hh   = (const float*)d_in[23];
  const float* Wn_dec  = (const float*)d_in[24];
  const float* bn_dec  = (const float*)d_in[25];
  float* out = (float*)d_out;

  // bf16 fragment-major weight scratch in d_ws (~288 KB)
  bf16_t* fih = (bf16_t*)d_ws;                 // [2][36][512]
  bf16_t* fhh = fih + 2 * 36 * 512;            // [6][36][512]

  srnn_prep_frag<<<144, 256, 0, stream>>>(Ws_ih, fih, 2);
  srnn_prep_frag<<<432, 256, 0, stream>>>(Ws_hh, fhh, 6);

  srnn_scan<<<NNODES, 256, 0, stream>>>(
      nodes, edges, hnode0, hedge0,
      Wt_enc, bt_enc, Wt_ih, Wt_hh, bt_ih, bt_hh,
      Ws_enc, bs_enc, bs_ih, bs_hh,
      Wn_enc, bn_enc, Wn_henc, bn_henc,
      Wn_ih, Wn_hh, bn_ih, bn_hh, Wn_dec, bn_dec,
      fih, fhh, out);
}
